// GPT2Model_68719476954
// MI455X (gfx1250) — matrix operand order repaired and checked by reading, not compiled, not run on hardware
//
#include <hip/hip_runtime.h>
#include <hip/hip_bf16.h>
#include <math.h>

// ---------------------------------------------------------------------------
// GPT-2 forward on gfx1250 (MI455X).
// ~1.24 TFLOP total vs ~500 MB compulsory traffic -> compute-bound; all GEMMs
// routed through v_wmma_f32_16x16x32_bf16 (fp32 accumulate), fp32->bf16 on the
// LDS staging path.  wave32.  GEMM: 128x128 tile, ping-pong LDS double buffer,
// 8 wmma/wave per K-step, global loads for the next tile in flight under the
// matrix math.  Attention: flash-style online softmax with WMMA for QK^T and
// P@KV.
// ---------------------------------------------------------------------------

#define Bq   4
#define Sq   1024
#define Eq   1024
#define Hq   16
#define Dq   64
#define Lq   12
#define E4q  4096
#define Mq   (Bq * Sq)          // 4096 rows
#define NEGV (-1e9f)

typedef __bf16 bf16_t;
typedef __attribute__((ext_vector_type(8)))  __bf16 v8bf;
typedef __attribute__((ext_vector_type(16))) __bf16 v16bf;
typedef __attribute__((ext_vector_type(8)))  float  v8f;

__device__ __forceinline__ v16bf cat8(v8bf lo, v8bf hi) {
  return __builtin_shufflevector(lo, hi, 0,1,2,3,4,5,6,7,8,9,10,11,12,13,14,15);
}
__device__ __forceinline__ v8f wmma_bf16(v16bf a, v16bf b, v8f c) {
  // (neg_a, A, neg_b, B, c_mod, C, reuse_a, reuse_b)
  return __builtin_amdgcn_wmma_f32_16x16x32_bf16(false, a, false, b, (short)0, c,
                                                 false, false);
}

// ---------------------------------------------------------------------------
// Embedding: h[b,s,:] = wte[x[b,s]] + wpe[s]
// ---------------------------------------------------------------------------
__global__ __launch_bounds__(256) void embed_kernel(
    const int* __restrict__ x, const float* __restrict__ wte,
    const float* __restrict__ wpe, float* __restrict__ h) {
  size_t i = (size_t)blockIdx.x * 256 + threadIdx.x;   // over B*S*E
  int e = (int)(i & (Eq - 1));
  size_t tokPos = i >> 10;                             // /E
  int s = (int)(tokPos & (Sq - 1));
  int tok = x[tokPos];
  h[i] = wte[(size_t)tok * Eq + e] + wpe[(size_t)s * Eq + e];
}

// ---------------------------------------------------------------------------
// Y[M,N] = act(X[M,K] @ W[K,N] + bias).  128x128 tile / 256-thread block.
// Wave (wm,wn) in a 4x2 grid owns a 32x64 subtile = 8 wmma accumulators.
// A staged row-major bf16; W staged transposed (Bt[n][k]) so both A and B
// fragments are contiguous ds_load_b128 reads in one shared k order for the bf16
// 16x32 (A) / 32x16 (B) operands, k = 16*half + s (not AMD's table order).  LDS ping-pong double buffer:
// one barrier per K-step, next tile's global loads issued before the wmma
// block so they overlap the matrix math.
// ---------------------------------------------------------------------------
template <bool GELU>
__global__ __launch_bounds__(256) void gemm_bias_kernel(
    const float* __restrict__ X, const float* __restrict__ W,
    const float* __restrict__ bias, float* __restrict__ Y,
    int N, int K) {
  __shared__ alignas(16) bf16_t As[2][128 * 32];   // [row][k]
  __shared__ alignas(16) bf16_t Bt[2][128 * 32];   // [n][k]

  const int tilesN = N >> 7;
  const int tm = blockIdx.x / tilesN, tn = blockIdx.x % tilesN;
  const int tid = threadIdx.x;
  const int lane = tid & 31, w = tid >> 5;
  const int wm = w & 3, wn = w >> 2;
  const int half = lane >> 4, ln16 = lane & 15;

  // staging coordinates: A 128x32 (16 floats/thread), B 32x128 (16/thread)
  const int ar = tid >> 1, ac = (tid & 1) * 16;
  const int bk = tid >> 3, bn = (tid & 7) * 16;
  const float* aSrc = X + (size_t)(tm * 128 + ar) * K + ac;
  const float* bSrc = W + (size_t)bk * N + tn * 128 + bn;

  v8f acc[2][4];
#pragma unroll
  for (int i = 0; i < 2; ++i)
#pragma unroll
    for (int j = 0; j < 4; ++j) acc[i][j] = v8f{};

  float aReg[16], bReg[16];
#pragma unroll
  for (int j = 0; j < 16; ++j) aReg[j] = aSrc[j];
#pragma unroll
  for (int j = 0; j < 16; ++j) bReg[j] = bSrc[j];
  {
    v8bf p0, p1;
#pragma unroll
    for (int j = 0; j < 8; ++j) { p0[j] = (bf16_t)aReg[j]; p1[j] = (bf16_t)aReg[8 + j]; }
    *(v8bf*)&As[0][ar * 32 + ac] = p0;
    *(v8bf*)&As[0][ar * 32 + ac + 8] = p1;
#pragma unroll
    for (int j = 0; j < 16; ++j) Bt[0][(bn + j) * 32 + bk] = (bf16_t)bReg[j];
  }
  __syncthreads();

  int cur = 0;
  for (int k0 = 0; k0 < K; k0 += 32, cur ^= 1) {
    const bool more = (k0 + 32) < K;
    if (more) {   // next tile's global loads: in flight under the wmma block
      const float* aN = aSrc + k0 + 32;
      const float* bN = bSrc + (size_t)(k0 + 32) * N;
#pragma unroll
      for (int j = 0; j < 16; ++j) aReg[j] = aN[j];
#pragma unroll
      for (int j = 0; j < 16; ++j) bReg[j] = bN[j];
    }

#pragma unroll
    for (int rs = 0; rs < 2; ++rs) {
      const bf16_t* arow = &As[cur][(wm * 32 + rs * 16 + ln16) * 32];
      v16bf af = cat8(*(const v8bf*)(arow + 16 * half),
                      *(const v8bf*)(arow + 16 * half + 8));
#pragma unroll
      for (int cs = 0; cs < 4; ++cs) {
        const bf16_t* bp = &Bt[cur][(wn * 64 + cs * 16 + ln16) * 32 + 16 * half];
        v16bf bf = cat8(((const v8bf*)bp)[0], ((const v8bf*)bp)[1]);
        acc[rs][cs] = wmma_bf16(af, bf, acc[rs][cs]);
      }
    }

    if (more) {
      const int nxt = cur ^ 1;
      v8bf p0, p1;
#pragma unroll
      for (int j = 0; j < 8; ++j) { p0[j] = (bf16_t)aReg[j]; p1[j] = (bf16_t)aReg[8 + j]; }
      *(v8bf*)&As[nxt][ar * 32 + ac] = p0;
      *(v8bf*)&As[nxt][ar * 32 + ac + 8] = p1;
#pragma unroll
      for (int j = 0; j < 16; ++j) Bt[nxt][(bn + j) * 32 + bk] = (bf16_t)bReg[j];
    }
    __syncthreads();
  }

  // --- epilogue: bias (+ exact GELU), fp32 store ---
#pragma unroll
  for (int cs = 0; cs < 4; ++cs) {
    int col = tn * 128 + wn * 64 + cs * 16 + ln16;
    float bv = bias[col];
#pragma unroll
    for (int rs = 0; rs < 2; ++rs) {
#pragma unroll
      for (int r = 0; r < 8; ++r) {
        int row = tm * 128 + wm * 32 + rs * 16 + 8 * half + r;
        float v = acc[rs][cs][r] + bv;
        if (GELU) v = 0.5f * v * (1.0f + erff(v * 0.70710678118654752f));
        Y[(size_t)row * N + col] = v;
      }
    }
  }
}

// ---------------------------------------------------------------------------
// Attention (per b,h): unscaled scores = Q KV^T, key-padding mask (x==0),
// softmax, out = P KV.  Flash-style: 64 queries/block (4 waves x 16 rows),
// stream key blocks of 64 with online softmax.  D = 64 -> 2 k-steps of 32.
// ---------------------------------------------------------------------------
__global__ __launch_bounds__(128) void attention_kernel(
    const float* __restrict__ q, const float* __restrict__ kv,
    const int* __restrict__ x, float* __restrict__ out) {
  __shared__ alignas(16) bf16_t Kv[64 * 64];    // [key][d]  (B-frags for QK^T)
  __shared__ alignas(16) bf16_t KvT[64 * 64];   // [d][key]  (B-frags for P@KV)
  __shared__ alignas(16) bf16_t Ps[4][16 * 64]; // per-wave P tile

  const int bid = blockIdx.x;
  const int qb = bid & 15;            // S/64 = 16 query blocks
  const int h  = (bid >> 4) & 15;
  const int b  = bid >> 8;
  const int tid = threadIdx.x, lane = tid & 31, w = tid >> 5;
  const int half = lane >> 4, ln16 = lane & 15;
  const int qRow0 = qb * 64 + w * 16;

  // Q fragments live in registers for the whole kernel (2 k-steps over D).
  v16bf qf[2];
  {
    const float* qrow = q + (size_t)(b * Sq + qRow0 + ln16) * Eq + h * 64;
#pragma unroll
    for (int s = 0; s < 2; ++s) {
      v16bf t;
#pragma unroll
      for (int j = 0; j < 8; ++j) t[j] = (bf16_t)qrow[32 * s + 16 * half + j];
#pragma unroll
      for (int j = 0; j < 8; ++j) t[8 + j] = (bf16_t)qrow[32 * s + 16 * half + 8 + j];
      qf[s] = t;
    }
  }

  float mrow[8], lrow[8];
  v8f o[4] = {v8f{}, v8f{}, v8f{}, v8f{}};
#pragma unroll
  for (int r = 0; r < 8; ++r) { mrow[r] = -3.0e38f; lrow[r] = 0.0f; }

  for (int kb = 0; kb < 16; ++kb) {
    // --- stage KV block (64 keys x 64 d) in both layouts ---
    {
      int r = tid >> 1, dOff = (tid & 1) * 32;
      const float* src = kv + (size_t)(b * Sq + kb * 64 + r) * Eq + h * 64 + dOff;
#pragma unroll
      for (int v4 = 0; v4 < 4; ++v4) {
        v8bf p;
#pragma unroll
        for (int j = 0; j < 8; ++j) p[j] = (bf16_t)src[v4 * 8 + j];
        *(v8bf*)&Kv[r * 64 + dOff + v4 * 8] = p;           // ds_store_b128
#pragma unroll
        for (int j = 0; j < 8; ++j) KvT[(dOff + v4 * 8 + j) * 64 + r] = p[j];
      }
    }
    __syncthreads();

    // --- scores: 16x64 via 4 wmma col-tiles x 2 k-steps ---
    v8f sc[4] = {v8f{}, v8f{}, v8f{}, v8f{}};
#pragma unroll
    for (int c = 0; c < 4; ++c) {
#pragma unroll
      for (int s = 0; s < 2; ++s) {
        const bf16_t* bp = &Kv[(c * 16 + ln16) * 64 + 32 * s + 16 * half];
        v16bf bfrag = cat8(((const v8bf*)bp)[0], ((const v8bf*)bp)[1]);
        sc[c] = wmma_bf16(qf[s], bfrag, sc[c]);
      }
    }

    // key-padding mask: this lane owns column ln16 of each 16-col tile
    bool msk[4];
#pragma unroll
    for (int c = 0; c < 4; ++c)
      msk[c] = (x[b * Sq + kb * 64 + c * 16 + ln16] == 0);

    // --- online softmax per row (rows 8*half + r, 16-lane reductions) ---
#pragma unroll
    for (int r = 0; r < 8; ++r) {
      float v0 = msk[0] ? NEGV : sc[0][r];
      float v1 = msk[1] ? NEGV : sc[1][r];
      float v2 = msk[2] ? NEGV : sc[2][r];
      float v3 = msk[3] ? NEGV : sc[3][r];
      float mx = fmaxf(fmaxf(v0, v1), fmaxf(v2, v3));
#pragma unroll
      for (int off = 8; off; off >>= 1) mx = fmaxf(mx, __shfl_xor(mx, off));
      float mnew = fmaxf(mrow[r], mx);
      float scale = __expf(mrow[r] - mnew);
      float e0 = __expf(v0 - mnew), e1 = __expf(v1 - mnew);
      float e2 = __expf(v2 - mnew), e3 = __expf(v3 - mnew);
      float rs = (e0 + e1) + (e2 + e3);
#pragma unroll
      for (int off = 8; off; off >>= 1) rs += __shfl_xor(rs, off);
      lrow[r] = lrow[r] * scale + rs;
      mrow[r] = mnew;
#pragma unroll
      for (int c = 0; c < 4; ++c) o[c][r] *= scale;
      int prow = 8 * half + r;
      Ps[w][prow * 64 +  0 + ln16] = (bf16_t)e0;
      Ps[w][prow * 64 + 16 + ln16] = (bf16_t)e1;
      Ps[w][prow * 64 + 32 + ln16] = (bf16_t)e2;
      Ps[w][prow * 64 + 48 + ln16] = (bf16_t)e3;
    }

    // --- O += P @ KV  (A-frags from per-wave P scratch, B from KvT) ---
#pragma unroll
    for (int s = 0; s < 2; ++s) {
      const bf16_t* prow = &Ps[w][ln16 * 64];
      v16bf af = cat8(*(const v8bf*)(prow + 32 * s + 16 * half),
                      *(const v8bf*)(prow + 32 * s + 16 * half + 8));
#pragma unroll
      for (int c = 0; c < 4; ++c) {
        const bf16_t* bp = &KvT[(c * 16 + ln16) * 64 + 32 * s + 16 * half];
        v16bf bfrag = cat8(((const v8bf*)bp)[0], ((const v8bf*)bp)[1]);
        o[c] = wmma_bf16(af, bfrag, o[c]);
      }
    }
    __syncthreads();
  }

  // --- normalize and write (B,S,E) layout at column h*64+d ---
#pragma unroll
  for (int c = 0; c < 4; ++c) {
    int col = h * 64 + c * 16 + ln16;
#pragma unroll
    for (int r = 0; r < 8; ++r) {
      int qrow = qRow0 + 8 * half + r;
      float denom = lrow[r];
      float v = (denom > 0.0f) ? o[c][r] / denom : 0.0f;
      out[(size_t)(b * Sq + qrow) * Eq + col] = v;
    }
  }
}

// ---------------------------------------------------------------------------
// out = layernorm(xin + res, g, beta); res may be null.  One block per row.
// ---------------------------------------------------------------------------
__global__ __launch_bounds__(256) void ln_res_kernel(
    const float* __restrict__ xin, const float* __restrict__ res,
    const float* __restrict__ g, const float* __restrict__ beta,
    float* __restrict__ out) {
  __shared__ float red[16];
  const int row = blockIdx.x;
  const float* xr = xin + (size_t)row * Eq;
  const float* rr = res ? res + (size_t)row * Eq : nullptr;
  float v[4], s = 0.0f, s2 = 0.0f;
#pragma unroll
  for (int j = 0; j < 4; ++j) {
    int i = threadIdx.x + j * 256;
    float t = xr[i] + (rr ? rr[i] : 0.0f);
    v[j] = t; s += t; s2 += t * t;
  }
#pragma unroll
  for (int off = 16; off; off >>= 1) {
    s  += __shfl_xor(s, off);
    s2 += __shfl_xor(s2, off);
  }
  int w = threadIdx.x >> 5, lane = threadIdx.x & 31;
  if (lane == 0) { red[w] = s; red[8 + w] = s2; }
  __syncthreads();
  if (w == 0) {
    float a  = (lane < 8) ? red[lane] : 0.0f;
    float b2 = (lane < 8) ? red[8 + lane] : 0.0f;
#pragma unroll
    for (int off = 4; off; off >>= 1) {
      a  += __shfl_xor(a, off);
      b2 += __shfl_xor(b2, off);
    }
    if (lane == 0) { red[0] = a; red[1] = b2; }
  }
  __syncthreads();
  float mean = red[0] * (1.0f / Eq);
  float var  = red[1] * (1.0f / Eq) - mean * mean;
  float inv  = rsqrtf(var + 1e-5f);
#pragma unroll
  for (int j = 0; j < 4; ++j) {
    int i = threadIdx.x + j * 256;
    out[(size_t)row * Eq + i] = (v[j] - mean) * inv * g[i] + beta[i];
  }
}

// ---------------------------------------------------------------------------
// Host orchestration.  Workspace layout (floats):
//   [0,4M)   h          [4M,8M)  q / u      [8M,12M) kv
//   [12M,16M) attn / m2 [16M,32M) fc        -> 128 MB total
// ---------------------------------------------------------------------------
extern "C" void kernel_launch(void* const* d_in, const int* in_sizes, int n_in,
                              void* d_out, int out_size, void* d_ws, size_t ws_size,
                              hipStream_t stream) {
  const int*   x     = (const int*)  d_in[0];
  const float* wte   = (const float*)d_in[1];
  const float* wpe   = (const float*)d_in[2];
  const float* Wq    = (const float*)d_in[3];
  const float* bq    = (const float*)d_in[4];
  const float* Wkv   = (const float*)d_in[5];
  const float* bkv   = (const float*)d_in[6];
  const float* Wfc   = (const float*)d_in[7];
  const float* bfc   = (const float*)d_in[8];
  const float* Wproj = (const float*)d_in[9];
  const float* bproj = (const float*)d_in[10];
  const float* ln1g  = (const float*)d_in[11];
  const float* ln1b  = (const float*)d_in[12];
  const float* ln2g  = (const float*)d_in[13];
  const float* ln2b  = (const float*)d_in[14];
  const float* lnfg  = (const float*)d_in[15];
  const float* lnfb  = (const float*)d_in[16];

  float* ws    = (float*)d_ws;
  const size_t T = (size_t)Mq * Eq;          // 4M floats
  float* h     = ws;
  float* qbuf  = ws + 1 * T;
  float* kvbuf = ws + 2 * T;
  float* atbuf = ws + 3 * T;
  float* fcbuf = ws + 4 * T;                 // 16M floats
  float* ubuf  = qbuf;                        // reuse (q dead after attention)
  float* m2buf = atbuf;                       // reuse (attn dead after ln1)

  embed_kernel<<<(Bq * Sq * Eq) / 256, 256, 0, stream>>>(x, wte, wpe, h);

  for (int l = 0; l < Lq; ++l) {
    const float* wq_l  = Wq    + (size_t)l * Eq * Eq;
    const float* bq_l  = bq    + (size_t)l * Eq;
    const float* wkv_l = Wkv   + (size_t)l * Eq * Eq;
    const float* bkv_l = bkv   + (size_t)l * Eq;
    const float* wfc_l = Wfc   + (size_t)l * Eq * E4q;
    const float* bfc_l = bfc   + (size_t)l * E4q;
    const float* wpj_l = Wproj + (size_t)l * E4q * Eq;
    const float* bpj_l = bproj + (size_t)l * Eq;

    gemm_bias_kernel<false><<<(Mq / 128) * (Eq / 128), 256, 0, stream>>>(
        h, wq_l, bq_l, qbuf, Eq, Eq);
    gemm_bias_kernel<false><<<(Mq / 128) * (Eq / 128), 256, 0, stream>>>(
        h, wkv_l, bkv_l, kvbuf, Eq, Eq);
    attention_kernel<<<Bq * Hq * (Sq / 64), 128, 0, stream>>>(
        qbuf, kvbuf, x, atbuf);
    ln_res_kernel<<<Mq, 256, 0, stream>>>(
        atbuf, h, ln1g + (size_t)l * Eq, ln1b + (size_t)l * Eq, ubuf);
    gemm_bias_kernel<true><<<(Mq / 128) * (E4q / 128), 256, 0, stream>>>(
        ubuf, wfc_l, bfc_l, fcbuf, E4q, Eq);
    gemm_bias_kernel<false><<<(Mq / 128) * (Eq / 128), 256, 0, stream>>>(
        fcbuf, wpj_l, bpj_l, m2buf, Eq, E4q);
    ln_res_kernel<<<Mq, 256, 0, stream>>>(
        m2buf, h, ln2g + (size_t)l * Eq, ln2b + (size_t)l * Eq, h);
  }

  ln_res_kernel<<<Mq, 256, 0, stream>>>(h, nullptr, lnfg, lnfb, (float*)d_out);
}
